// Model_42803644072527
// MI455X (gfx1250) — matrix operand order repaired and checked by reading, not compiled, not run on hardware
//
#include <hip/hip_runtime.h>

// ---------------------------------------------------------------------------
// Types for CDNA5 WMMA (wave32): 16x16x32 bf16 -> f32
// ---------------------------------------------------------------------------
typedef __attribute__((ext_vector_type(16))) __bf16 v16bf;
typedef __attribute__((ext_vector_type(8)))  float  v8f;

// ---------------------------------------------------------------------------
// Helpers
// ---------------------------------------------------------------------------
__device__ __forceinline__ unsigned short f2bf(float f) {
  unsigned u = __float_as_uint(f);
  unsigned r = u + 0x7fffu + ((u >> 16) & 1u);                 // RNE
  if ((u & 0x7fffffffu) > 0x7f800000u) r = u | 0x00400000u;    // quiet NaN
  return (unsigned short)(r >> 16);
}

// Order-preserving float<->uint mapping so atomicMax(uint) == float max
__device__ __forceinline__ unsigned ford(float f) {
  unsigned u = __float_as_uint(f);
  return (u & 0x80000000u) ? ~u : (u | 0x80000000u);
}
__device__ __forceinline__ float unford(unsigned e) {
  unsigned u = (e & 0x80000000u) ? (e & 0x7fffffffu) : ~e;
  return __uint_as_float(u);
}
#define ENC_NEG_INF 0x007FFFFFu   // ford(-inf)

__device__ __forceinline__ float lrelu(float v) { return v > 0.f ? v : 0.2f * v; }

__device__ __forceinline__ void edge_sd(const int* __restrict__ ei, int e, int E,
                                        int* s, int* d) {
  if (e < E) { *s = ei[e]; *d = ei[E + e]; }
  else       { *s = e - E; *d = e - E; }       // self loop
}

// ---------------------------------------------------------------------------
// fp32 -> bf16 conversion (activations, row-major A operand)
// ---------------------------------------------------------------------------
__global__ void k_f32_to_bf16(const float* __restrict__ in,
                              unsigned short* __restrict__ out, int n) {
  int i = blockIdx.x * blockDim.x + threadIdx.x;
  if (i < n) out[i] = f2bf(in[i]);
}

// ---------------------------------------------------------------------------
// Pack W[K=128, cin] fp32 -> native WMMA B-fragments (bf16), zero pad to cout.
// Fragment layout: Bp[(((colTile*4 + kt)*32) + lane)*16 + 2p + j]
//   = W[(kt*32 + (p<4 ? 2p : 16+2(p-4)) + 8*(lane>>4) + j) * cin + colTile*16 + (lane&15)]
// (B operand: same K order as A: pairs at K = {0,2,4,6,16,18,20,22} + 8*half)
// ---------------------------------------------------------------------------
__global__ void k_pack_w(const float* __restrict__ W, unsigned short* __restrict__ Bp,
                         int cin, int cout) {
  int i = blockIdx.x * blockDim.x + threadIdx.x;
  if (i >= 128 * cout) return;
  int elem = i & 15;
  int lane = (i >> 4) & 31;
  int kt   = (i >> 9) & 3;          // K/32 = 4 k-tiles
  int colTile = i >> 11;
  int p = elem >> 1, j = elem & 1;
  int kk = kt * 32 + ((p < 4) ? (2 * p) : (16 + 2 * (p - 4))) + 8 * (lane >> 4) + j;
  int c  = colTile * 16 + (lane & 15);
  Bp[i] = (c < cin) ? f2bf(W[kk * cin + c]) : (unsigned short)0;
}

// ---------------------------------------------------------------------------
// GEMM: D[N,C] = A[N,128](bf16) * W (packed B-fragments), f32 accumulate.
// One wave per 16x16 output tile; block = 4 waves; grid = (N/16, C/64).
// K=128 hardcoded -> fully unrolled: 4x v_wmma_f32_16x16x32_bf16 per wave.
// ---------------------------------------------------------------------------
__global__ void k_gemm_wmma(const __bf16* __restrict__ A,
                            const __bf16* __restrict__ Bp,
                            float* __restrict__ D, int C) {
  const int K = 128;
  int lane = threadIdx.x & 31;
  int wave = threadIdx.x >> 5;
  int half = lane >> 4;
  int m16  = lane & 15;
  long rowBase = (long)blockIdx.x * 16;
  int  colTile = blockIdx.y * 4 + wave;

  const __bf16* arow  = A + (rowBase + m16) * K;
  const __bf16* bfrag = Bp + ((long)colTile * 4 * 32 + lane) * 16;

  v8f acc = {};
#pragma unroll
  for (int kt = 0; kt < 4; ++kt) {
    v16bf a;
#pragma unroll
    for (int p = 0; p < 8; ++p) {
      // ISA 16-bit A 16x32 layout: pairs at K = {0,2,4,6,16,18,20,22} + 8*half
      int kb = ((p < 4) ? (2 * p) : (16 + 2 * (p - 4))) + 8 * half + kt * 32;
      a[2 * p]     = arow[kb];
      a[2 * p + 1] = arow[kb + 1];
    }
    v16bf b = *(const v16bf*)(bfrag + (long)kt * 32 * 16);   // one 32B fragment load
    acc = __builtin_amdgcn_wmma_f32_16x16x32_bf16(
        /*neg_a=*/false, a, /*neg_b=*/false, b,
        /*c_mod=*/(short)0, acc, /*reuse_a=*/false, /*reuse_b=*/false);
  }
#pragma unroll
  for (int v = 0; v < 8; ++v) {
    long row = rowBase + v + 8 * half;           // D layout: VGPR v -> M=v / M=8+v
    D[row * C + colTile * 16 + m16] = acc[v];
  }
}

// ---------------------------------------------------------------------------
// Per-node attention coefficients
// ---------------------------------------------------------------------------
// C=128, 4 heads x 32 ch: one wave per node, lane -> 4 channels, 8-lane head groups
__global__ void k_alpha128(const float* __restrict__ H,
                           const float* __restrict__ as_, const float* __restrict__ ad_,
                           float* __restrict__ aS, float* __restrict__ aD, int n) {
  int gw = (blockIdx.x * blockDim.x + threadIdx.x) >> 5;
  int lane = threadIdx.x & 31;
  if (gw >= n) return;
  int c = lane * 4;
  int head = lane >> 3;
  int hc = c & 31;
  const float4 hv = *(const float4*)(H + (long)gw * 128 + c);
  const float4 sv = *(const float4*)(as_ + head * 32 + hc);
  const float4 dv = *(const float4*)(ad_ + head * 32 + hc);
  float ps = hv.x * sv.x + hv.y * sv.y + hv.z * sv.z + hv.w * sv.w;
  float pd = hv.x * dv.x + hv.y * dv.y + hv.z * dv.z + hv.w * dv.w;
#pragma unroll
  for (int m = 1; m < 8; m <<= 1) {
    ps += __shfl_xor(ps, m, 8);
    pd += __shfl_xor(pd, m, 8);
  }
  if ((lane & 7) == 0) {
    aS[gw * 4 + head] = ps;
    aD[gw * 4 + head] = pd;
  }
}

// 1 head x 40 ch (H stride 64): one wave per node, full-wave reduce
__global__ void k_alpha40(const float* __restrict__ H,
                          const float* __restrict__ as_, const float* __restrict__ ad_,
                          float* __restrict__ aS, float* __restrict__ aD, int n) {
  int gw = (blockIdx.x * blockDim.x + threadIdx.x) >> 5;
  int lane = threadIdx.x & 31;
  if (gw >= n) return;
  const float* h = H + (long)gw * 64;
  float ps = h[lane] * as_[lane];                 // lane < 32 <= 40 always valid
  float pd = h[lane] * ad_[lane];
  if (lane + 32 < 40) {
    ps += h[lane + 32] * as_[lane + 32];
    pd += h[lane + 32] * ad_[lane + 32];
  }
#pragma unroll
  for (int m = 1; m < 32; m <<= 1) {
    ps += __shfl_xor(ps, m, 32);
    pd += __shfl_xor(pd, m, 32);
  }
  if (lane == 0) { aS[gw] = ps; aD[gw] = pd; }
}

// ---------------------------------------------------------------------------
// Segment softmax over edges (3 passes), wave-per-edge aggregation
// ---------------------------------------------------------------------------
__global__ void k_fill_u32(unsigned* __restrict__ p, unsigned v, int n) {
  int i = blockIdx.x * blockDim.x + threadIdx.x;
  if (i < n) p[i] = v;
}

__global__ void k_edge_max(const int* __restrict__ ei, int E, int Etot, int heads,
                           const float* __restrict__ aS, const float* __restrict__ aD,
                           unsigned* __restrict__ emax) {
  int e = blockIdx.x * blockDim.x + threadIdx.x;
  if (e >= Etot) return;
  int s, d; edge_sd(ei, e, E, &s, &d);
  for (int h = 0; h < heads; ++h) {
    float v = lrelu(aS[s * heads + h] + aD[d * heads + h]);
    atomicMax(&emax[d * heads + h], ford(v));
  }
}

__global__ void k_edge_expsum(const int* __restrict__ ei, int E, int Etot, int heads,
                              const float* __restrict__ aS, const float* __restrict__ aD,
                              const unsigned* __restrict__ emax,
                              float* __restrict__ denom) {
  int e = blockIdx.x * blockDim.x + threadIdx.x;
  if (e >= Etot) return;
  int s, d; edge_sd(ei, e, E, &s, &d);
  for (int h = 0; h < heads; ++h) {
    float v = lrelu(aS[s * heads + h] + aD[d * heads + h]);
    atomicAdd(&denom[d * heads + h], __expf(v - unford(emax[d * heads + h])));
  }
}

// wave per edge; lane -> 4 consecutive channels (head = lane>>3); float4 gather
__global__ void k_agg128(const int* __restrict__ ei, int E, int Etot,
                         const float* __restrict__ aS, const float* __restrict__ aD,
                         const unsigned* __restrict__ emax, const float* __restrict__ denom,
                         const float* __restrict__ H, float* __restrict__ out) {
  int gw = (blockIdx.x * blockDim.x + threadIdx.x) >> 5;
  int lane = threadIdx.x & 31;
  if (gw >= Etot) return;
  int s, d; edge_sd(ei, gw, E, &s, &d);
  int head = lane >> 3;
  float v = lrelu(aS[s * 4 + head] + aD[d * 4 + head]);
  float ex = __expf(v - unford(emax[d * 4 + head]));
  float alpha = ex / (denom[d * 4 + head] + 1e-16f);
  const float4 hv = *(const float4*)(H + (long)s * 128 + lane * 4);
  float* o = out + (long)d * 128 + lane * 4;
  atomicAdd(o + 0, hv.x * alpha);
  atomicAdd(o + 1, hv.y * alpha);
  atomicAdd(o + 2, hv.z * alpha);
  atomicAdd(o + 3, hv.w * alpha);
}

// heads=1, ch=40, H stride 64, out stride 40
__global__ void k_agg40(const int* __restrict__ ei, int E, int Etot,
                        const float* __restrict__ aS, const float* __restrict__ aD,
                        const unsigned* __restrict__ emax, const float* __restrict__ denom,
                        const float* __restrict__ H, float* __restrict__ out) {
  int gw = (blockIdx.x * blockDim.x + threadIdx.x) >> 5;
  int lane = threadIdx.x & 31;
  if (gw >= Etot) return;
  int s, d; edge_sd(ei, gw, E, &s, &d);
  float v = lrelu(aS[s] + aD[d]);
  float ex = __expf(v - unford(emax[d]));
  float alpha = ex / (denom[d] + 1e-16f);
  const float* h = H + (long)s * 64;
  float* o = out + (long)d * 40;
  atomicAdd(o + lane, h[lane] * alpha);
  if (lane + 32 < 40) atomicAdd(o + lane + 32, h[lane + 32] * alpha);
}

// ---------------------------------------------------------------------------
// BatchNorm (population stats over N) + ReLU, fused with bf16 convert
// ---------------------------------------------------------------------------
__global__ void k_bn_stats(const float* __restrict__ X, float* __restrict__ sum,
                           float* __restrict__ sumsq, int n) {
  __shared__ float ls[128], lq[128];
  if (threadIdx.x < 128) { ls[threadIdx.x] = 0.f; lq[threadIdx.x] = 0.f; }
  __syncthreads();
  int rowsPer = (n + gridDim.x - 1) / gridDim.x;
  long r0 = (long)blockIdx.x * rowsPer;
  long r1 = r0 + rowsPer; if (r1 > n) r1 = n;
  float s = 0.f, q = 0.f;
  for (long idx = r0 * 128 + threadIdx.x; idx < r1 * 128; idx += 256) {
    float v = X[idx]; s += v; q += v * v;
  }
  int c = threadIdx.x & 127;
  atomicAdd(&ls[c], s);
  atomicAdd(&lq[c], q);
  __syncthreads();
  if (threadIdx.x < 128) {
    atomicAdd(&sum[threadIdx.x],   ls[threadIdx.x]);
    atomicAdd(&sumsq[threadIdx.x], lq[threadIdx.x]);
  }
}

__global__ void k_bn_finalize(const float* __restrict__ sum, const float* __restrict__ sumsq,
                              const float* __restrict__ g, const float* __restrict__ be,
                              float* __restrict__ scale, float* __restrict__ shift, int n) {
  int c = threadIdx.x;
  float inv = 1.f / (float)n;
  float mu  = sum[c] * inv;
  float var = sumsq[c] * inv - mu * mu;
  float sc  = rsqrtf(var + 1e-5f) * g[c];
  scale[c] = sc;
  shift[c] = be[c] - mu * sc;
}

__global__ void k_bn_apply(const float* __restrict__ X, const float* __restrict__ scale,
                           const float* __restrict__ shift,
                           unsigned short* __restrict__ Xbf, int total) {
  int i = blockIdx.x * blockDim.x + threadIdx.x;
  if (i >= total) return;
  int c = i & 127;
  float f = X[i] * scale[c] + shift[c];
  f = f > 0.f ? f : 0.f;          // ReLU
  Xbf[i] = f2bf(f);
}

// ---------------------------------------------------------------------------
// log_softmax over 40 logits, wave per node
// ---------------------------------------------------------------------------
__global__ void k_logsoftmax40(const float* __restrict__ X, const float* __restrict__ bias,
                               float* __restrict__ out, int n) {
  int gw = (blockIdx.x * blockDim.x + threadIdx.x) >> 5;
  int lane = threadIdx.x & 31;
  if (gw >= n) return;
  const float* row = X + (long)gw * 40;
  float v0 = row[lane] + bias[lane];                         // lane < 32 <= 40
  bool hi = (lane + 32 < 40);
  float v1 = hi ? (row[lane + 32] + bias[lane + 32]) : -3.4e38f;
  float m = fmaxf(v0, v1);
#pragma unroll
  for (int mk = 1; mk < 32; mk <<= 1) m = fmaxf(m, __shfl_xor(m, mk, 32));
  float s = __expf(v0 - m) + (hi ? __expf(v1 - m) : 0.f);
#pragma unroll
  for (int mk = 1; mk < 32; mk <<= 1) s += __shfl_xor(s, mk, 32);
  float lg = __logf(s);
  out[(long)gw * 40 + lane] = v0 - m - lg;
  if (hi) out[(long)gw * 40 + lane + 32] = v1 - m - lg;
}

// ---------------------------------------------------------------------------
// Host orchestration
// ---------------------------------------------------------------------------
extern "C" void kernel_launch(void* const* d_in, const int* in_sizes, int n_in,
                              void* d_out, int out_size, void* d_ws, size_t ws_size,
                              hipStream_t stream) {
  const float* x   = (const float*)d_in[0];
  const int*   ei  = (const int*)d_in[1];
  const float* Ws[2]  = {(const float*)d_in[2],  (const float*)d_in[8]};
  const float* ass[2] = {(const float*)d_in[3],  (const float*)d_in[9]};
  const float* ads[2] = {(const float*)d_in[4],  (const float*)d_in[10]};
  const float* gs[2]  = {(const float*)d_in[6],  (const float*)d_in[12]};
  const float* bes[2] = {(const float*)d_in[7],  (const float*)d_in[13]};
  const float* W2  = (const float*)d_in[14];
  const float* as2 = (const float*)d_in[15];
  const float* ad2 = (const float*)d_in[16];
  const float* b2  = (const float*)d_in[17];

  const int N = in_sizes[0] / 128;
  const int E = in_sizes[1] / 2;
  const int Etot = E + N;           // with self loops

  char* p = (char*)d_ws;
  size_t off = 0;
  auto take = [&](size_t b) { void* r = p + off; off = (off + b + 255) & ~(size_t)255; return r; };
  unsigned short* Xbf  = (unsigned short*)take((size_t)N * 128 * 2);
  float*          Hf   = (float*)take((size_t)N * 128 * 4);
  float*          AGG  = (float*)take((size_t)N * 128 * 4);
  float*          aS   = (float*)take((size_t)N * 4 * 4);
  float*          aD   = (float*)take((size_t)N * 4 * 4);
  unsigned*       emax = (unsigned*)take((size_t)N * 4 * 4);
  float*          den  = (float*)take((size_t)N * 4 * 4);
  unsigned short* Wb   = (unsigned short*)take(128 * 128 * 2);  // packed B-fragments
  float*          sums = (float*)take(128 * 4);
  float*          sq   = (float*)take(128 * 4);
  float*          scl  = (float*)take(128 * 4);
  float*          sft  = (float*)take(128 * 4);

  const int T = 256;
  k_f32_to_bf16<<<(N * 128 + T - 1) / T, T, 0, stream>>>(x, Xbf, N * 128);

  // ---- Layers 0 & 1: GAT(128->4x32) + BN + ReLU ----
  for (int l = 0; l < 2; ++l) {
    k_pack_w<<<(128 * 128 + T - 1) / T, T, 0, stream>>>(Ws[l], Wb, 128, 128);
    k_gemm_wmma<<<dim3(N / 16, 2), 128, 0, stream>>>(
        (const __bf16*)Xbf, (const __bf16*)Wb, Hf, 128);
    k_alpha128<<<(N + 7) / 8, T, 0, stream>>>(Hf, ass[l], ads[l], aS, aD, N);

    hipMemsetAsync(den, 0, (size_t)N * 4 * 4, stream);
    k_fill_u32<<<(N * 4 + T - 1) / T, T, 0, stream>>>(emax, ENC_NEG_INF, N * 4);
    k_edge_max<<<(Etot + T - 1) / T, T, 0, stream>>>(ei, E, Etot, 4, aS, aD, emax);
    k_edge_expsum<<<(Etot + T - 1) / T, T, 0, stream>>>(ei, E, Etot, 4, aS, aD, emax, den);

    hipMemsetAsync(AGG, 0, (size_t)N * 128 * 4, stream);
    k_agg128<<<(Etot + 7) / 8, T, 0, stream>>>(ei, E, Etot, aS, aD, emax, den, Hf, AGG);

    hipMemsetAsync(sums, 0, 128 * 4, stream);
    hipMemsetAsync(sq,   0, 128 * 4, stream);
    k_bn_stats<<<1024, 256, 0, stream>>>(AGG, sums, sq, N);
    k_bn_finalize<<<1, 128, 0, stream>>>(sums, sq, gs[l], bes[l], scl, sft, N);
    k_bn_apply<<<(N * 128 + T - 1) / T, T, 0, stream>>>(AGG, scl, sft, Xbf, N * 128);
  }

  // ---- Layer 2: GAT(128->1x40), W zero-padded to 64 cols for WMMA tiling ----
  k_pack_w<<<(128 * 64 + T - 1) / T, T, 0, stream>>>(W2, Wb, 40, 64);
  k_gemm_wmma<<<dim3(N / 16, 1), 128, 0, stream>>>(
      (const __bf16*)Xbf, (const __bf16*)Wb, Hf, 64);
  k_alpha40<<<(N + 7) / 8, T, 0, stream>>>(Hf, as2, ad2, aS, aD, N);

  hipMemsetAsync(den, 0, (size_t)N * 4, stream);
  k_fill_u32<<<(N + T - 1) / T, T, 0, stream>>>(emax, ENC_NEG_INF, N);
  k_edge_max<<<(Etot + T - 1) / T, T, 0, stream>>>(ei, E, Etot, 1, aS, aD, emax);
  k_edge_expsum<<<(Etot + T - 1) / T, T, 0, stream>>>(ei, E, Etot, 1, aS, aD, emax, den);

  hipMemsetAsync(AGG, 0, (size_t)N * 40 * 4, stream);
  k_agg40<<<(Etot + 7) / 8, T, 0, stream>>>(ei, E, Etot, aS, aD, emax, den, Hf, AGG);

  k_logsoftmax40<<<(N + 7) / 8, T, 0, stream>>>(AGG, b2, (float*)d_out, N);
}
